// BasicLayer_62577673503407
// MI455X (gfx1250) — matrix operand order repaired and checked by reading, not compiled, not run on hardware
//
#include <hip/hip_runtime.h>

// ---------------------------------------------------------------------------
// Fused windowed-GCN layer for MI455X (gfx1250, wave32, WMMA bf16)
// One workgroup (8 wave32) per (batch, output-timestep) window.
// ---------------------------------------------------------------------------

#define BQ 8
#define TQ 64
#define NQ 128
#define DQ 128
#define TOUT 62          // T - 2

typedef __attribute__((ext_vector_type(16))) __bf16 v16bf;
typedef __attribute__((ext_vector_type(8)))  __bf16 v8bf;
typedef __attribute__((ext_vector_type(4)))  __bf16 v4bf;
typedef __attribute__((ext_vector_type(8)))  float  v8f;

// LDS strides in elements.
// WT_S: 392 elems = 784 B = 196 dwords; 196 mod 64 = 4 -> unit-d lane step hits
// banks 4 apart (<=2-way conflict), and 784 is a multiple of 16 B (b128 loads ok).
#define WT_S 392   // transposed 128 x 384 bf16 arrays (whT, yT)
#define RM_S 136   // 128 x 128 bf16 arrays (272 B rows, 16B-aligned)
#define SB_S 132   // 128 x 128 f32 array

// LDS byte offsets (dynamic shared memory)
#define OFF_WHT 0                       // 128*392*2 = 100352
#define OFF_YT  100352                  // 128*392*2 = 100352
#define OFF_WHS 200704                  // 128*136*2 = 34816
#define OFF_KT  235520                  // 128*136*2 = 34816
#define OFF_S   270336                  // 128 f32
#define OFF_DIS (OFF_S + 512)           // 384 f32
#define SMEM_BYTES (OFF_DIS + 1536)     // 272384 B (< 320 KB WGP LDS)
// aliased regions (live ranges do not overlap; barriers separate phases)
#define OFF_AGG 0                       // over dead whT
#define OFF_W1T 34816                   // over dead whT
#define OFF_W2T 100352                  // over dead yT
#define OFF_H1  135168                  // over dead yT
#define OFF_SB  200704                  // over dead whS+KT (67584 <= 69632)

__device__ __forceinline__ v8f wmma_bf16(v16bf a, v16bf b, v8f c) {
  return __builtin_amdgcn_wmma_f32_16x16x32_bf16(
      /*neg_a=*/false, a, /*neg_b=*/false, b,
      /*c_mod=*/(short)0, c, /*reuse_a=*/false, /*reuse_b=*/false);
}

// A operand (16x32 MxK, bf16): lane m = L&15; VGPR0-3 hold K = 8h+2v(+1),
// VGPR4-7 hold K = 16+8h+2v(+1)  -> two contiguous 8-elem runs per lane.
__device__ __forceinline__ v16bf load_a(const __bf16* base, int stride,
                                        int m0, int k0, int lane) {
  int m = m0 + (lane & 15);
  int h = lane >> 4;
  const __bf16* p = base + m * stride + k0 + h * 8;
  union { v16bf v; v8bf h8[2]; } u;
  u.h8[0] = *(const v8bf*)p;
  u.h8[1] = *(const v8bf*)(p + 16);
  return u.v;
}

// B operand (32x16 KxN, bf16): lane n = L&15; K = 8h+0..7, then 16+8h+0..7.
// `base` holds the matrix TRANSPOSED: base[n*stride + k].
__device__ __forceinline__ v16bf load_b(const __bf16* base, int stride,
                                        int n0, int k0, int lane) {
  int n = n0 + (lane & 15);
  int h = lane >> 4;
  const __bf16* p = base + n * stride + k0 + h * 8;
  union { v16bf v; v8bf h8[2]; } u;
  u.h8[0] = *(const v8bf*)p;
  u.h8[1] = *(const v8bf*)(p + 16);
  return u.v;
}

__global__ void __launch_bounds__(256, 1)
fused_gcn_win_kernel(const float* __restrict__ feat, const float* __restrict__ w,
                     const float* __restrict__ W1, const float* __restrict__ b1,
                     const float* __restrict__ W2, const float* __restrict__ b2,
                     const float* __restrict__ gamma, const float* __restrict__ beta,
                     float* __restrict__ out)
{
  extern __shared__ __align__(16) char smem[];
  __bf16* whT  = (__bf16*)(smem + OFF_WHT);  // [128][WT_S]  whT[d][n] = wh[n][d]
  __bf16* yT   = (__bf16*)(smem + OFF_YT);   // [128][WT_S]  yT[e][n]  = dis[n]*win[n][e]
  __bf16* whS  = (__bf16*)(smem + OFF_WHS);  // [128][RM_S]  wh rows 256..383 row-major
  __bf16* KT   = (__bf16*)(smem + OFF_KT);   // [128][RM_S]  KT[e][d] = K[d][e]
  float*  Sv   = (float*)(smem + OFF_S);     // [128]
  float*  dis  = (float*)(smem + OFF_DIS);   // [384]
  __bf16* aggB = (__bf16*)(smem + OFF_AGG);  // [128][RM_S]
  __bf16* W1T  = (__bf16*)(smem + OFF_W1T);  // [128][RM_S]  W1T[j][e] = W1[e][j]
  __bf16* W2T  = (__bf16*)(smem + OFF_W2T);  // [128][RM_S]
  __bf16* h1B  = (__bf16*)(smem + OFF_H1);   // [128][RM_S]
  float*  sB   = (float*)(smem + OFF_SB);    // [128][SB_S]

  const int tid  = threadIdx.x;
  const int lane = tid & 31;
  const int wv   = tid >> 5;                 // wave id 0..7
  const int bq   = blockIdx.x / TOUT;
  const int tq   = blockIdx.x % TOUT;
  const size_t g0 = (((size_t)bq * TQ) + tq) * NQ * DQ;   // window rows 0..383 contiguous

  const v8f vzero = {0.f,0.f,0.f,0.f,0.f,0.f,0.f,0.f};

  // per-lane sigmoid gate; lane owns features d = lane + 32q
  float sg[4];
  #pragma unroll
  for (int q = 0; q < 4; ++q) {
    int d = lane + 32 * q;
    sg[q] = 1.0f / (1.0f + expf(-w[d]));
  }

  // ---- Phase 1: load window, gate, L2-normalize rows.
  // 4 rows per group so the transposed writes pack into b64 stores
  // (4 consecutive r at fixed d).
  for (int grp = 0; grp < 12; ++grp) {
    int r0 = wv * 48 + grp * 4;
    float vr[4][4];   // [row][q] raw
    float hv[4][4];   // [row][q] normalized gated
    #pragma unroll
    for (int rr = 0; rr < 4; ++rr) {
      int r = r0 + rr;
      float ss = 0.f;
      #pragma unroll
      for (int q = 0; q < 4; ++q) {
        float x = feat[g0 + (size_t)r * DQ + lane + 32 * q];   // coalesced b32
        vr[rr][q] = x;
        float gx = x * sg[q];
        hv[rr][q] = gx;
        ss += gx * gx;
      }
      #pragma unroll
      for (int mk = 16; mk >= 1; mk >>= 1) ss += __shfl_xor(ss, mk, 32);
      float inv = 1.0f / fmaxf(sqrtf(ss), 1e-12f);
      #pragma unroll
      for (int q = 0; q < 4; ++q) hv[rr][q] *= inv;
    }
    #pragma unroll
    for (int q = 0; q < 4; ++q) {
      int d = lane + 32 * q;
      v4bf wp, yp;
      #pragma unroll
      for (int rr = 0; rr < 4; ++rr) {
        wp[rr] = (__bf16)hv[rr][q];
        yp[rr] = (__bf16)vr[rr][q];
      }
      *(v4bf*)(whT + d * WT_S + r0) = wp;    // one b64 store
      *(v4bf*)(yT  + d * WT_S + r0) = yp;    // one b64 store
    }
    if (r0 >= 256) {
      #pragma unroll
      for (int rr = 0; rr < 4; ++rr) {
        int rws = r0 + rr - 256;
        #pragma unroll
        for (int q = 0; q < 4; ++q)
          whS[rws * RM_S + lane + 32 * q] = (__bf16)hv[rr][q];
      }
    }
  }
  __syncthreads();

  // ---- Phase 2: column sum S[d] = sum_n wh[n][d] (contiguous row of whT)
  if (tid < 128) {
    float s = 0.f;
    for (int n = 0; n < 384; n += 8) {
      v8bf c = *(const v8bf*)(whT + tid * WT_S + n);
      #pragma unroll
      for (int q = 0; q < 8; ++q) s += (float)c[q];
    }
    Sv[tid] = s;
  }
  __syncthreads();

  // ---- Phase 3: deg_m = <wh_m, S>;  dis_m = deg>0 ? rsqrt(deg) : 0
  for (int m = tid; m < 384; m += 256) {
    float deg = 0.f;
    for (int d = 0; d < 128; ++d) deg += (float)whT[d * WT_S + m] * Sv[d];
    dis[m] = (deg > 0.f) ? rsqrtf(deg) : 0.f;
  }
  __syncthreads();

  // ---- Phase 4a: scale yT columns by dis[n]  (Y = D_n * Win), b64 RMW chunks
  {
    int d  = tid >> 1;
    int nb = (tid & 1) * 192;
    for (int n = nb; n < nb + 192; n += 4) {
      v4bf t = *(v4bf*)(yT + d * WT_S + n);
      v4bf o;
      #pragma unroll
      for (int q = 0; q < 4; ++q) o[q] = (__bf16)((float)t[q] * dis[n + q]);
      *(v4bf*)(yT + d * WT_S + n) = o;
    }
  }
  __syncthreads();

  // ---- Phase 4: K = Wh^T * Y   (128x128, k-depth 384) -> KT (transposed bf16)
  {
    const int md = wv;
    v8f acc[8];
    #pragma unroll
    for (int e = 0; e < 8; ++e) acc[e] = vzero;
    for (int kb = 0; kb < 12; ++kb) {
      v16bf a = load_a(whT, WT_S, md * 16, kb * 32, lane);
      #pragma unroll
      for (int eb = 0; eb < 8; ++eb) {
        v16bf bm = load_b(yT, WT_S, eb * 16, kb * 32, lane);
        acc[eb] = wmma_bf16(a, bm, acc[eb]);
      }
    }
    int half = lane >> 4;
    #pragma unroll
    for (int eb = 0; eb < 8; ++eb) {
      int e = eb * 16 + (lane & 15);
      int dbase = md * 16 + half * 8;
      v8bf st;
      #pragma unroll
      for (int r = 0; r < 8; ++r) st[r] = (__bf16)acc[eb][r];
      *(v8bf*)(KT + e * RM_S + dbase) = st;
    }
  }
  __syncthreads();

  // ---- Phase 5: agg = D_m * (Wh_sub @ K)  -> aggB (bf16 row-major)
  {
    const int md = wv;
    v8f acc[8];
    #pragma unroll
    for (int e = 0; e < 8; ++e) acc[e] = vzero;
    #pragma unroll
    for (int kb = 0; kb < 4; ++kb) {
      v16bf a = load_a(whS, RM_S, md * 16, kb * 32, lane);
      #pragma unroll
      for (int eb = 0; eb < 8; ++eb) {
        v16bf bm = load_b(KT, RM_S, eb * 16, kb * 32, lane);
        acc[eb] = wmma_bf16(a, bm, acc[eb]);
      }
    }
    int half = lane >> 4;
    #pragma unroll
    for (int eb = 0; eb < 8; ++eb) {
      int e = eb * 16 + (lane & 15);
      #pragma unroll
      for (int r = 0; r < 8; ++r) {
        int m = md * 16 + half * 8 + r;
        aggB[m * RM_S + e] = (__bf16)(acc[eb][r] * dis[256 + m]);
      }
    }
  }
  // cooperative transposed weight staging (whT/yT regions now dead)
  for (int idx = tid; idx < 16384; idx += 256) {
    int e = idx >> 7, j = idx & 127;          // idx = e*128 + j (coalesced global)
    W1T[j * RM_S + e] = (__bf16)W1[idx];
    W2T[j * RM_S + e] = (__bf16)W2[idx];
  }
  __syncthreads();

  // ---- Phase 6: h1 = relu(agg @ W1 + b1) -> h1B bf16
  {
    const int md = wv;
    v8f acc[8];
    #pragma unroll
    for (int e = 0; e < 8; ++e) acc[e] = vzero;
    #pragma unroll
    for (int kb = 0; kb < 4; ++kb) {
      v16bf a = load_a(aggB, RM_S, md * 16, kb * 32, lane);
      #pragma unroll
      for (int eb = 0; eb < 8; ++eb) {
        v16bf bm = load_b(W1T, RM_S, eb * 16, kb * 32, lane);
        acc[eb] = wmma_bf16(a, bm, acc[eb]);
      }
    }
    int half = lane >> 4;
    #pragma unroll
    for (int eb = 0; eb < 8; ++eb) {
      int j = eb * 16 + (lane & 15);
      float bj = b1[j];
      #pragma unroll
      for (int r = 0; r < 8; ++r) {
        int m = md * 16 + half * 8 + r;
        h1B[m * RM_S + j] = (__bf16)fmaxf(acc[eb][r] + bj, 0.f);
      }
    }
  }
  __syncthreads();

  // ---- Phase 7: s = feat[:,t+2] + h1 @ W2 + b2 -> sB (f32)
  {
    const int md = wv;
    const size_t gres = (((size_t)bq * TQ) + tq + 2) * NQ * DQ;
    v8f acc[8];
    #pragma unroll
    for (int e = 0; e < 8; ++e) acc[e] = vzero;
    #pragma unroll
    for (int kb = 0; kb < 4; ++kb) {
      v16bf a = load_a(h1B, RM_S, md * 16, kb * 32, lane);
      #pragma unroll
      for (int eb = 0; eb < 8; ++eb) {
        v16bf bm = load_b(W2T, RM_S, eb * 16, kb * 32, lane);
        acc[eb] = wmma_bf16(a, bm, acc[eb]);
      }
    }
    int half = lane >> 4;
    #pragma unroll
    for (int eb = 0; eb < 8; ++eb) {
      int o = eb * 16 + (lane & 15);
      float bo = b2[o];
      #pragma unroll
      for (int r = 0; r < 8; ++r) {
        int m = md * 16 + half * 8 + r;
        sB[m * SB_S + o] = acc[eb][r] + bo + feat[gres + (size_t)m * DQ + o];
      }
    }
  }
  __syncthreads();

  // ---- Phase 8: LayerNorm over D + writeout (each wave owns 16 rows)
  {
    const size_t obase = (((size_t)bq * TOUT) + tq) * NQ * DQ;
    for (int i = 0; i < 16; ++i) {
      int m = wv * 16 + i;
      float4 xv = *(const float4*)(sB + m * SB_S + lane * 4);
      float s1 = xv.x + xv.y + xv.z + xv.w;
      float s2 = xv.x*xv.x + xv.y*xv.y + xv.z*xv.z + xv.w*xv.w;
      #pragma unroll
      for (int mk = 16; mk >= 1; mk >>= 1) {
        s1 += __shfl_xor(s1, mk, 32);
        s2 += __shfl_xor(s2, mk, 32);
      }
      float mu   = s1 * (1.0f / 128.0f);
      float var  = s2 * (1.0f / 128.0f) - mu * mu;
      float rstd = rsqrtf(var + 1e-5f);
      float xs[4] = {xv.x, xv.y, xv.z, xv.w};
      #pragma unroll
      for (int q = 0; q < 4; ++q) {
        int d = lane * 4 + q;
        out[obase + (size_t)m * DQ + d] = (xs[q] - mu) * rstd * gamma[d] + beta[d];
      }
    }
  }
}

extern "C" void kernel_launch(void* const* d_in, const int* in_sizes, int n_in,
                              void* d_out, int out_size, void* d_ws, size_t ws_size,
                              hipStream_t stream) {
  (void)in_sizes; (void)n_in; (void)d_ws; (void)ws_size; (void)out_size;
  const float* feat  = (const float*)d_in[0];
  const float* w     = (const float*)d_in[1];
  const float* W1    = (const float*)d_in[2];
  const float* b1    = (const float*)d_in[3];
  const float* W2    = (const float*)d_in[4];
  const float* b2    = (const float*)d_in[5];
  const float* gamma = (const float*)d_in[6];
  const float* beta  = (const float*)d_in[7];
  float* out = (float*)d_out;

  dim3 grid(BQ * TOUT);   // 496 windows
  dim3 block(256);        // 8 wave32 per workgroup
  fused_gcn_win_kernel<<<grid, block, SMEM_BYTES, stream>>>(
      feat, w, W1, b1, W2, b2, gamma, beta, out);
}
